// MomAttention_13752485282222
// MI455X (gfx1250) — matrix operand order repaired and checked by reading, not compiled, not run on hardware
//
#include <hip/hip_runtime.h>
#include <hip/hip_bf16.h>

typedef __attribute__((ext_vector_type(16))) __bf16 v16bf;
typedef __attribute__((ext_vector_type(8)))  __bf16 v8bf;
typedef __attribute__((ext_vector_type(8)))  float  v8f;
typedef __attribute__((ext_vector_type(4)))  int    v4i;

#define B_    4
#define S_    2048
#define D_    1024
#define H_    8
#define M_    4
#define DK_   128
#define T_    (B_ * S_)    // 8192 tokens
#define VS_   64           // v-columns per scan workgroup
#define KSTEP 64           // K-tile (2 WMMA K-depths)

// ---------------------------------------------------------------------------
// CDNA5 async global->LDS copy (ASYNCcnt-tracked), with portable fallback.
// Probe round 2 showed param0 is an AS1 (global) v4i*, so: (global, lds, off, cpol).
// ---------------------------------------------------------------------------
#if __has_builtin(__builtin_amdgcn_global_load_async_to_lds_b128)
#define USE_ASYNC_LDS 1
#endif

typedef __attribute__((address_space(1))) v4i gv4i;
typedef __attribute__((address_space(3))) v4i lv4i;

__device__ __forceinline__ void async_cp16(void* lds, const void* glob) {
#ifdef USE_ASYNC_LDS
  __builtin_amdgcn_global_load_async_to_lds_b128(
      (gv4i*)(uintptr_t)glob, (lv4i*)(unsigned)(uintptr_t)lds, 0, 0);
#else
  *(float4*)lds = *(const float4*)glob;
#endif
}

__device__ __forceinline__ void async_wait0() {
#ifdef USE_ASYNC_LDS
#if __has_builtin(__builtin_amdgcn_s_wait_asynccnt)
  __builtin_amdgcn_s_wait_asynccnt(0);
#else
  asm volatile("s_wait_asynccnt 0" ::: "memory");
#endif
#endif
}

// ---------------------------------------------------------------------------
// Precision prep: x -> bf16 copy; W -> bf16 transposed [N][D] copy.
// ---------------------------------------------------------------------------
__global__ __launch_bounds__(256)
void cvt_bf16_kernel(const float* __restrict__ src, __bf16* __restrict__ dst, int n4) {
  int i = (blockIdx.x * 256 + threadIdx.x);
  if (i < n4) {
    float4 v = ((const float4*)src)[i];
    __bf16* p = dst + (size_t)i * 4;
    p[0] = (__bf16)v.x; p[1] = (__bf16)v.y; p[2] = (__bf16)v.z; p[3] = (__bf16)v.w;
  }
}

__global__ __launch_bounds__(256)
void transpose_bf16_kernel(const float* __restrict__ W, __bf16* __restrict__ Wt,
                           int rows, int cols) {  // W[rows][cols] -> Wt[cols][rows]
  __shared__ float tile[32][33];
  const int bx = blockIdx.x * 32;   // col tile
  const int by = blockIdx.y * 32;   // row tile
  const int tx = threadIdx.x & 31;
  const int ty = threadIdx.x >> 5;  // 0..7
#pragma unroll
  for (int i = 0; i < 32; i += 8)
    tile[ty + i][tx] = W[(size_t)(by + ty + i) * cols + bx + tx];
  __syncthreads();
#pragma unroll
  for (int i = 0; i < 32; i += 8)
    Wt[(size_t)(bx + ty + i) * rows + by + tx] = (__bf16)tile[tx][ty + i];
}

// ---------------------------------------------------------------------------
// bf16 WMMA GEMM: Y[T_,N] = act(X[T_,D_] @ Wt^T);  Wt is [N][D_] bf16.
// 128x128 tile/WG, 8 waves (4x2), each wave 32x64 (2x4 wmma tiles), K-tile 64,
// double-buffered LDS fed by async global->LDS b128 copies.
// ---------------------------------------------------------------------------
template <int ACT>
__global__ __launch_bounds__(256)
void gemm_bf16_wmma(const __bf16* __restrict__ X, const __bf16* __restrict__ Wt,
                    float* __restrict__ Y, int N) {
  __shared__ alignas(32) __bf16 lA[2][128 * KSTEP];   // [row][k]
  __shared__ alignas(32) __bf16 lB[2][128 * KSTEP];   // [n][k]

  const int tid  = threadIdx.x;
  const int lane = tid & 31;
  const int wave = tid >> 5;
  const int wm   = wave & 3;
  const int wn   = wave >> 2;
  const int t0   = blockIdx.x * 128;
  const int n0   = blockIdx.y * 128;

  const v8f vzero = {0.f, 0.f, 0.f, 0.f, 0.f, 0.f, 0.f, 0.f};
  v8f acc[2][4];
#pragma unroll
  for (int i = 0; i < 2; ++i)
#pragma unroll
    for (int j = 0; j < 4; ++j) acc[i][j] = vzero;

  // Each thread copies a 64B run (4 x b128) per matrix per tile.
  const int cr  = tid >> 1;             // row / n : 0..127
  const int cc0 = (tid & 1) * 32;       // bf16 col offset: 0 or 32

  auto issueTile = [&](int buf, int kb) {
    const __bf16* ga = X  + (size_t)(t0 + cr) * D_ + kb + cc0;
    const __bf16* gb = Wt + (size_t)(n0 + cr) * D_ + kb + cc0;
    __bf16* la = &lA[buf][cr * KSTEP + cc0];
    __bf16* lb = &lB[buf][cr * KSTEP + cc0];
#pragma unroll
    for (int u = 0; u < 4; ++u) {
      async_cp16(la + u * 8, ga + u * 8);
      async_cp16(lb + u * 8, gb + u * 8);
    }
  };

  auto compute = [&](int buf) {
#pragma unroll
    for (int c = 0; c < KSTEP; c += 32) {
      v16bf afr[2];
#pragma unroll
      for (int i = 0; i < 2; ++i) {
        const int row = wm * 32 + i * 16 + (lane & 15);
        const int kbl = (lane < 16) ? 0 : 8;
        const __bf16* ap = &lA[buf][row * KSTEP + c + kbl];
        v8bf lo = *(const v8bf*)ap;
        v8bf hi = *(const v8bf*)(ap + 16);
        afr[i] = __builtin_shufflevector(lo, hi, 0, 1, 2, 3, 4, 5, 6, 7,
                                         8, 9, 10, 11, 12, 13, 14, 15);
      }
#pragma unroll
      for (int j = 0; j < 4; ++j) {
        const int n   = wn * 64 + j * 16 + (lane & 15);
        const int kbb = (lane < 16) ? 0 : 8;
        const __bf16* bp = &lB[buf][n * KSTEP + c + kbb];
        v8bf blo = *(const v8bf*)bp;
        v8bf bhi = *(const v8bf*)(bp + 16);
        v16bf bfr = __builtin_shufflevector(blo, bhi, 0, 1, 2, 3, 4, 5, 6, 7,
                                            8, 9, 10, 11, 12, 13, 14, 15);
#pragma unroll
        for (int i = 0; i < 2; ++i)
          acc[i][j] = __builtin_amdgcn_wmma_f32_16x16x32_bf16(
              false, afr[i], false, bfr, (short)0, acc[i][j], false, false);
      }
    }
  };

  issueTile(0, 0);
  async_wait0();
  __syncthreads();
  int buf = 0;
  for (int kb = 0; kb < D_; kb += KSTEP) {
    if (kb + KSTEP < D_) issueTile(buf ^ 1, kb + KSTEP);
    compute(buf);
    async_wait0();
    __syncthreads();
    buf ^= 1;
  }

#pragma unroll
  for (int i = 0; i < 2; ++i)
#pragma unroll
    for (int j = 0; j < 4; ++j)
#pragma unroll
      for (int r = 0; r < 8; ++r) {
        const int row = t0 + wm * 32 + i * 16 + ((lane < 16) ? r : r + 8);
        const int col = n0 + wn * 64 + j * 16 + (lane & 15);
        float val = acc[i][j][r];
        if (ACT == 1) val = val / (1.f + __expf(-val));   // silu
        Y[(size_t)row * N + col] = val;
      }
}

// ---------------------------------------------------------------------------
// Small projections: g = logsigmoid(x@Wg), beta = sigmoid(x@Wb),
// rw = renormalized top-2 softmax(x@Wr), masked. One block per token.
// ---------------------------------------------------------------------------
__global__ __launch_bounds__(128)
void proj_small_kernel(const float* __restrict__ x, const unsigned char* __restrict__ am,
                       const float* __restrict__ Wg, const float* __restrict__ Wb,
                       const float* __restrict__ Wr, float* __restrict__ g,
                       float* __restrict__ beta, float* __restrict__ rw) {
  const int t = blockIdx.x;
  const int tid = threadIdx.x;
  float acc[20];
#pragma unroll
  for (int o = 0; o < 20; ++o) acc[o] = 0.f;
  for (int d = tid; d < D_; d += 128) {
    const float xv = x[(size_t)t * D_ + d];
    const float* wg = Wg + (size_t)d * H_;
    const float* wb = Wb + (size_t)d * H_;
    const float* wr = Wr + (size_t)d * M_;
#pragma unroll
    for (int o = 0; o < 8; ++o) acc[o] += xv * wg[o];
#pragma unroll
    for (int o = 0; o < 8; ++o) acc[8 + o] += xv * wb[o];
#pragma unroll
    for (int o = 0; o < 4; ++o) acc[16 + o] += xv * wr[o];
  }
  __shared__ float red[20][128];
#pragma unroll
  for (int o = 0; o < 20; ++o) red[o][tid] = acc[o];
  __syncthreads();
  for (int s = 64; s > 0; s >>= 1) {
    if (tid < s)
#pragma unroll
      for (int o = 0; o < 20; ++o) red[o][tid] += red[o][tid + s];
    __syncthreads();
  }
  if (tid == 0) {
    const float amv = am[t] ? 1.f : 0.f;
#pragma unroll
    for (int hh = 0; hh < 8; ++hh) {
      const float zg = red[hh][0];
      g[t * H_ + hh] = (zg < 0.f) ? (zg - log1pf(__expf(zg))) : (-log1pf(__expf(-zg)));
      const float zb = red[8 + hh][0];
      beta[t * H_ + hh] = 1.f / (1.f + __expf(-zb));
    }
    float p[4], mx = red[16][0];
#pragma unroll
    for (int mm = 1; mm < 4; ++mm) mx = fmaxf(mx, red[16 + mm][0]);
    float sum = 0.f;
#pragma unroll
    for (int mm = 0; mm < 4; ++mm) { p[mm] = __expf(red[16 + mm][0] - mx); sum += p[mm]; }
#pragma unroll
    for (int mm = 0; mm < 4; ++mm) p[mm] /= sum;
    int i1 = 0;
#pragma unroll
    for (int mm = 1; mm < 4; ++mm) if (p[mm] > p[i1]) i1 = mm;
    int i2 = (i1 == 0) ? 1 : 0;
#pragma unroll
    for (int mm = 0; mm < 4; ++mm) if (mm != i1 && p[mm] > p[i2]) i2 = mm;
    const float s2 = p[i1] + p[i2];
#pragma unroll
    for (int mm = 0; mm < 4; ++mm) {
      float rv = (mm == i1) ? p[i1] / s2 : (mm == i2) ? p[i2] / s2 : 0.f;
      rw[t * M_ + mm] = rv * amv;
    }
  }
}

// ---------------------------------------------------------------------------
// L2-normalize each k head-row of 128 (one wave32 per row).
// ---------------------------------------------------------------------------
__global__ __launch_bounds__(256)
void knorm_kernel(float* __restrict__ k) {
  const int row  = blockIdx.x * 8 + (threadIdx.x >> 5);
  const int lane = threadIdx.x & 31;
  float4* p = (float4*)(k + (size_t)row * DK_) + lane;
  float4 val = *p;
  float ss = val.x * val.x + val.y * val.y + val.z * val.z + val.w * val.w;
#pragma unroll
  for (int off = 16; off > 0; off >>= 1) ss += __shfl_xor(ss, off, 32);
  const float inv = 1.f / (sqrtf(ss) + 1e-6f);
  val.x *= inv; val.y *= inv; val.z *= inv; val.w *= inv;
  *p = val;
}

__global__ void zerof_kernel(float* __restrict__ p, int n) {
  int i = blockIdx.x * 256 + threadIdx.x;
  if (i < n) p[i] = 0.f;
}

// ---------------------------------------------------------------------------
// Gated delta-rule scan: per-(m,h,b,v-half) 128x64 f32 state LDS-resident.
// ---------------------------------------------------------------------------
__global__ __launch_bounds__(256)
void scan_kernel(const float* __restrict__ q, const float* __restrict__ k,
                 const float* __restrict__ v, const float* __restrict__ g,
                 const float* __restrict__ beta, const float* __restrict__ rw,
                 float* __restrict__ omix) {
  __shared__ float St[128 * VS_];
  __shared__ float qv[128], kv[128], vv[VS_], dlt[VS_];
  __shared__ float red[256];
  __shared__ float sc[3];

  const int tid = threadIdx.x;
  const int bid = blockIdx.x;
  const int vh = bid & 1;
  const int m  = (bid >> 1) & 3;
  const int h  = (bid >> 3) & 7;
  const int b  = bid >> 6;
  const int vbase = vh * VS_;

  for (int i = tid; i < 128 * VS_; i += 256) St[i] = 0.f;
  __syncthreads();

  const int vc = tid & (VS_ - 1);
  const int kq = tid >> 6;

  for (int s = 0; s < S_; ++s) {
    const int t = b * S_ + s;
    const size_t base = (size_t)t * (H_ * DK_) + h * DK_;
    if (tid < 128) {
      qv[tid] = q[base + tid];
      kv[tid] = k[base + tid];
    } else if (tid < 128 + VS_) {
      vv[tid - 128] = v[base + vbase + (tid - 128)];
    } else if (tid == 192) {
      const float rwv = rw[t * M_ + m];
      const float msk = rwv > 0.f ? 1.f : 0.f;
      sc[0] = (msk > 0.f) ? __expf(g[t * H_ + h]) : 1.f;
      sc[1] = beta[t * H_ + h] * msk;
      sc[2] = rwv;
    }
    __syncthreads();
    const float decay = sc[0], bm = sc[1], w = sc[2];

    for (int i = tid; i < 128 * VS_; i += 256) St[i] *= decay;
    __syncthreads();

    {   // pred[v] = sum_k kv[k] * St[k][v]
      float ps = 0.f;
#pragma unroll 8
      for (int kk = kq * 32; kk < kq * 32 + 32; ++kk) ps += kv[kk] * St[kk * VS_ + vc];
      red[tid] = ps;
      __syncthreads();
      if (tid < VS_)
        dlt[tid] = (vv[tid] - (red[tid] + red[tid + 64] + red[tid + 128] + red[tid + 192])) * bm;
      __syncthreads();
    }

    {   // St[k][v] += kv[k] * dlt[v]
      const float dv = dlt[vc];
#pragma unroll 8
      for (int kk = kq * 32; kk < kq * 32 + 32; ++kk) St[kk * VS_ + vc] += kv[kk] * dv;
    }
    __syncthreads();

    {   // o[v] = sum_k qv[k] * St[k][v]
      float ps = 0.f;
#pragma unroll 8
      for (int kk = kq * 32; kk < kq * 32 + 32; ++kk) ps += qv[kk] * St[kk * VS_ + vc];
      red[tid] = ps;
      __syncthreads();
      if (tid < VS_ && w > 0.f)
        atomicAdd(&omix[base + vbase + tid],
                  w * (red[tid] + red[tid + 64] + red[tid + 128] + red[tid + 192]));
      __syncthreads();
    }
  }
}

// ---------------------------------------------------------------------------
extern "C" void kernel_launch(void* const* d_in, const int* in_sizes, int n_in,
                              void* d_out, int out_size, void* d_ws, size_t ws_size,
                              hipStream_t stream) {
  const float*         x  = (const float*)d_in[0];
  const unsigned char* am = (const unsigned char*)d_in[1];
  const float*         Wq = (const float*)d_in[2];
  const float*         Wk = (const float*)d_in[3];
  const float*         Wv = (const float*)d_in[4];
  const float*         Wg = (const float*)d_in[5];
  const float*         Wb = (const float*)d_in[6];
  const float*         Wr = (const float*)d_in[7];
  const float*         Wo = (const float*)d_in[8];
  float*               out = (float*)d_out;

  const size_t TD = (size_t)T_ * D_;         // 8388608
  char* ws = (char*)d_ws;
  size_t off = 0;
  auto alloc = [&](size_t bytes) { char* p = ws + off; off += (bytes + 255) & ~(size_t)255; return p; };

  float*  q     = (float*)alloc(TD * 4);
  float*  k     = (float*)alloc(TD * 4);
  float*  v     = (float*)alloc(TD * 4);
  float*  gbuf  = (float*)alloc((size_t)T_ * H_ * 4);
  float*  bbuf  = (float*)alloc((size_t)T_ * H_ * 4);
  float*  rwb   = (float*)alloc((size_t)T_ * M_ * 4);
  float*  omix  = (float*)alloc(TD * 4);
  __bf16* xb    = (__bf16*)alloc(TD * 2);
  __bf16* omixb = (__bf16*)alloc(TD * 2);
  __bf16* Wqb   = (__bf16*)alloc((size_t)D_ * D_ * 2);
  __bf16* Wkb   = (__bf16*)alloc((size_t)D_ * D_ * 2);
  __bf16* Wvb   = (__bf16*)alloc((size_t)D_ * D_ * 2);
  __bf16* Wob   = (__bf16*)alloc((size_t)D_ * D_ * 2);

  // 1) precision prep (one-time per call): x -> bf16, W -> bf16 transposed [N][D]
  cvt_bf16_kernel<<<(int)(TD / 4 + 255) / 256, 256, 0, stream>>>(x, xb, (int)(TD / 4));
  dim3 tg(D_ / 32, D_ / 32);
  transpose_bf16_kernel<<<tg, 256, 0, stream>>>(Wq, Wqb, D_, D_);
  transpose_bf16_kernel<<<tg, 256, 0, stream>>>(Wk, Wkb, D_, D_);
  transpose_bf16_kernel<<<tg, 256, 0, stream>>>(Wv, Wvb, D_, D_);
  transpose_bf16_kernel<<<tg, 256, 0, stream>>>(Wo, Wob, D_, D_);

  // 2) big projections on the WMMA path
  dim3 gdim(T_ / 128, D_ / 128);
  gemm_bf16_wmma<1><<<gdim, 256, 0, stream>>>(xb, Wqb, q, D_);   // silu(x@Wq)
  gemm_bf16_wmma<1><<<gdim, 256, 0, stream>>>(xb, Wkb, k, D_);   // silu(x@Wk)
  gemm_bf16_wmma<0><<<gdim, 256, 0, stream>>>(xb, Wvb, v, D_);   // x@Wv

  // 3) gates / routing / k-norm
  proj_small_kernel<<<T_, 128, 0, stream>>>(x, am, Wg, Wb, Wr, gbuf, bbuf, rwb);
  knorm_kernel<<<(T_ * H_) / 8, 256, 0, stream>>>(k);

  // 4) sequential gated delta-rule scan (LDS-resident state)
  zerof_kernel<<<(int)((TD + 255) / 256), 256, 0, stream>>>(omix, (int)TD);
  scan_kernel<<<B_ * H_ * M_ * 2, 256, 0, stream>>>(q, k, v, gbuf, bbuf, rwb, omix);

  // 5) output projection on the WMMA path
  cvt_bf16_kernel<<<(int)(TD / 4 + 255) / 256, 256, 0, stream>>>(omix, omixb, (int)(TD / 4));
  gemm_bf16_wmma<0><<<gdim, 256, 0, stream>>>(omixb, Wob, out, D_);
}
